// LiteDGCNN_42709154791522
// MI455X (gfx1250) — matrix operand order repaired and checked by reading, not compiled, not run on hardware
//
#include <hip/hip_runtime.h>
#include <hip/hip_bf16.h>

// ---------------------------------------------------------------------------
// LiteDGCNN forward for MI455X (gfx1250, wave32, WMMA).
// All GEMMs use v_wmma_f32_16x16x32_bf16 (f32 accumulate).
// Training-mode BN => stats pass + apply pass per BN layer, chained on stream.
// B-matrices are swizzled into fragment-major LDS layout so each WMMA operand
// is 2x ds_load_b128; fragments are loaded per-use from LDS (register-hoisting
// them spilled to scratch in round 2 -- LDS reload is the right choice here).
// ---------------------------------------------------------------------------

typedef __bf16 bf16;
typedef bf16  v16bf __attribute__((ext_vector_type(16)));
typedef bf16  v8bf  __attribute__((ext_vector_type(8)));
typedef float v8f   __attribute__((ext_vector_type(8)));

#define EPS_   1e-5f
#define B_     16
#define N_     2048
#define KNN_   20
#define C1_    64
#define C2_    128
#define EMB_   1024
#define OUT_   7
#define NEDGE_ (B_*N_*KNN_)   // 655360 edges
#define NROW_  (B_*N_)        // 32768 nodes

// A-fragment loader: 16x32 bf16 A tile, row-major row pointer.
// Lane half 'hi' covers K {16hi..16hi+15} within k-step 'ks'.
// Two aligned 16B runs -> 2x ds_load_b128.
__device__ __forceinline__ v16bf load_afrag(const bf16* __restrict__ row, int hi, int ks) {
  const int base = 32 * ks + 16 * hi;
  v8bf lo = *(const v8bf*)(row + base);
  v8bf hv = *(const v8bf*)(row + base + 8);
  return __builtin_shufflevector(lo, hv, 0,1,2,3,4,5,6,7,8,9,10,11,12,13,14,15);
}

// ---------------------------------------------------------------------------
// Kernel 1: per-cloud kNN (squared Euclidean, self included).
// ---------------------------------------------------------------------------
__global__ void knn_kernel(const float* __restrict__ pos, int* __restrict__ idx) {
  __shared__ float sp[N_ * 3];
  const int b = blockIdx.y;
  const int i = blockIdx.x * 256 + threadIdx.x;
  const float* p = pos + (size_t)b * N_ * 3;
  for (int t = threadIdx.x; t < N_ * 3; t += 256) sp[t] = p[t];
  __syncthreads();
  const float px = sp[i*3+0], py = sp[i*3+1], pz = sp[i*3+2];
  float bd[KNN_]; int bi[KNN_];
#pragma unroll
  for (int t = 0; t < KNN_; ++t) { bd[t] = 3.4e38f; bi[t] = i; }
  for (int j = 0; j < N_; ++j) {
    float dx = px - sp[j*3+0], dy = py - sp[j*3+1], dz = pz - sp[j*3+2];
    float d = dx*dx + dy*dy + dz*dz;
    if (d < bd[KNN_-1]) {
      float cd = d; int ci = j;
#pragma unroll
      for (int t = 0; t < KNN_; ++t) {
        if (cd < bd[t]) { float td=bd[t]; int ti=bi[t]; bd[t]=cd; bi[t]=ci; cd=td; ci=ti; }
      }
    }
  }
  int* o = idx + (size_t)(b * N_ + i) * KNN_;
#pragma unroll
  for (int t = 0; t < KNN_; ++t) o[t] = bi[t];
}

// ---------------------------------------------------------------------------
// Kernel 2: layer-1 pre-activation stats (z1 = e@W1 + b1, 64 ch). VALU (K=6).
// ---------------------------------------------------------------------------
__global__ void l1_stats_kernel(const float* __restrict__ pos, const int* __restrict__ idx,
                                const float* __restrict__ W1, const float* __restrict__ b1,
                                float* __restrict__ stats1) {
  __shared__ float feat[64][6];
  __shared__ float w1[6 * C1_];
  __shared__ float red[2][256];
  const int base = blockIdx.x * 64;
  const int t = threadIdx.x;
  for (int q = t; q < 6 * C1_; q += 256) w1[q] = W1[q];
  for (int q = t; q < 64 * 6; q += 256) {
    int el = q / 6, f = q % 6;
    int edge = base + el;
    int nf = edge / KNN_;
    int b  = nf >> 11;
    int j  = idx[edge];
    int d  = (f < 3) ? f : f - 3;
    float vi = pos[(size_t)nf * 3 + d];
    feat[el][f] = (f < 3) ? vi : (pos[((size_t)(b * N_) + j) * 3 + d] - vi);
  }
  __syncthreads();
  const int c = t & 63, slice = t >> 6;
  float w[6];
#pragma unroll
  for (int f = 0; f < 6; ++f) w[f] = w1[f * C1_ + c];
  const float bb = b1[c];
  float s = 0.f, ss = 0.f;
  for (int e = slice * 16; e < slice * 16 + 16; ++e) {
    float z = bb;
#pragma unroll
    for (int f = 0; f < 6; ++f) z += feat[e][f] * w[f];
    s += z; ss += z * z;
  }
  red[0][t] = s; red[1][t] = ss;
  __syncthreads();
  if (t < 64) {
    float S = 0.f, SS = 0.f;
    for (int sl = 0; sl < 4; ++sl) { S += red[0][sl*64 + t]; SS += red[1][sl*64 + t]; }
    atomicAdd(&stats1[t], S);
    atomicAdd(&stats1[64 + t], SS);
  }
}

// ---------------------------------------------------------------------------
// Kernel 3: BN finalize.
// ---------------------------------------------------------------------------
__global__ void bn_finalize_kernel(const float* __restrict__ stats,
                                   const float* __restrict__ g, const float* __restrict__ be,
                                   float* __restrict__ scale, float* __restrict__ shift,
                                   int nch, float invCnt) {
  int c = blockIdx.x * blockDim.x + threadIdx.x;
  if (c >= nch) return;
  float m = stats[c] * invCnt;
  float v = stats[nch + c] * invCnt - m * m;
  float s = g[c] * rsqrtf(v + EPS_);
  scale[c] = s;
  shift[c] = be[c] - m * s;
}

// ---------------------------------------------------------------------------
// Kernel 4: fused edge MLP with WMMA.
// B fragments read per-use from fragment-major LDS (2x ds_load_b128 each).
// MODE 0: layer-2 stats.  MODE 1: BN2+ReLU + max over 20 neighbors.
// ---------------------------------------------------------------------------
template <int MODE>
__global__ void __launch_bounds__(128)
edge_mlp_kernel(const float* __restrict__ pos, const int* __restrict__ idx,
                const float* __restrict__ W1, const float* __restrict__ b1,
                const float* __restrict__ sc1, const float* __restrict__ sh1,
                const float* __restrict__ W2, const float* __restrict__ b2,
                const float* __restrict__ sc2, const float* __restrict__ sh2,
                float* __restrict__ stats2, float* __restrict__ node_out) {
  // W2 in fragment-major order: frag i = nt*2+ks; element [i][lane][e] contiguous.
  __shared__ v16bf sw2v[16 * 32];                       // 16KB
  __shared__ bf16  sh1buf[4][16 * C1_] __attribute__((aligned(16)));  // 8KB
  __shared__ float sw1[6 * C1_];
  __shared__ float sb1[C1_], ssc1[C1_], ssh1[C1_];
  __shared__ float sb2[C2_], ssc2[C2_], ssh2[C2_];
  __shared__ float ssum[C2_], sss[C2_];
  __shared__ unsigned snode[16 * C2_];                  // 8KB

  const int tid = threadIdx.x;                          // blockDim = 128 (4 waves)
  {
    bf16* sw2b = (bf16*)sw2v;
    for (int q = tid; q < C1_ * C2_; q += 128) {
      int k = q >> 7, col = q & 127;
      int ks = k >> 5, rem = k & 31, h2 = rem >> 4, e = rem & 15;
      int nt = col >> 4, n = col & 15;
      int L = h2 * 16 + n;
      sw2b[(((nt * 2 + ks) * 32 + L) << 4) + e] = (bf16)W2[q];
    }
  }
  for (int q = tid; q < 6 * C1_; q += 128) sw1[q] = W1[q];
  if (tid < C1_) { sb1[tid] = b1[tid]; ssc1[tid] = sc1[tid]; ssh1[tid] = sh1[tid]; }
  if (tid < C2_) {
    sb2[tid] = b2[tid];
    if (MODE == 1) { ssc2[tid] = sc2[tid]; ssh2[tid] = sh2[tid]; }
    if (MODE == 0) { ssum[tid] = 0.f; sss[tid] = 0.f; }
  }
  if (MODE == 1) for (int q = tid; q < 16 * C2_; q += 128) snode[q] = 0u;
  __syncthreads();

  const int w = tid >> 5, lane = tid & 31;
  const int hi = lane >> 4, r = lane & 15;
  const int nodeBase = blockIdx.x * 16;
  const long edgeBase = (long)nodeBase * KNN_;

  for (int tile = w; tile < 20; tile += 4) {
    // ---- compute h1 for row r, channels [32*hi, 32*hi+32) ----
    const int  eLocal = tile * 16 + r;
    const long edge   = edgeBase + eLocal;
    const int  nf     = (int)(edge / KNN_);
    const int  b      = nf >> 11;
    const int  j      = idx[edge];
    const float e0 = pos[(size_t)nf*3+0], e1 = pos[(size_t)nf*3+1], e2 = pos[(size_t)nf*3+2];
    const size_t pj = ((size_t)(b * N_) + j) * 3;
    const float e3 = pos[pj+0]-e0, e4 = pos[pj+1]-e1, e5 = pos[pj+2]-e2;
    const int ch0 = hi * 32;
#pragma unroll 4
    for (int c = 0; c < 32; ++c) {
      int cg = ch0 + c;
      float z = sb1[cg] + e0*sw1[cg] + e1*sw1[64+cg] + e2*sw1[128+cg]
                        + e3*sw1[192+cg] + e4*sw1[256+cg] + e5*sw1[320+cg];
      z = z * ssc1[cg] + ssh1[cg];
      sh1buf[w][r * C1_ + cg] = (bf16)fmaxf(z, 0.f);
    }
    // intra-wave LDS RAW fence (DS ops in-order per wave; keep compiler honest)
    asm volatile("s_wait_dscnt 0" ::: "memory");

    const bf16* arow = &sh1buf[w][r * C1_];
    v16bf a0 = load_afrag(arow, hi, 0);
    v16bf a1 = load_afrag(arow, hi, 1);

#pragma unroll
    for (int nt = 0; nt < 8; ++nt) {
      const int col = nt * 16 + r;
      v16bf bb0 = sw2v[(nt * 2 + 0) * 32 + lane];
      v16bf bb1 = sw2v[(nt * 2 + 1) * 32 + lane];
      v8f acc = {};
      acc = __builtin_amdgcn_wmma_f32_16x16x32_bf16(false, a0, false, bb0, (short)0, acc, false, false);
      acc = __builtin_amdgcn_wmma_f32_16x16x32_bf16(false, a1, false, bb1, (short)0, acc, false, false);
      const float bc = sb2[col];
      if (MODE == 0) {
        float s = 0.f, ss = 0.f;
#pragma unroll
        for (int v = 0; v < 8; ++v) { float z = acc[v] + bc; s += z; ss += z * z; }
        atomicAdd(&ssum[col], s);
        atomicAdd(&sss[col], ss);
      } else {
        const float sc = ssc2[col], sh = ssh2[col];
#pragma unroll
        for (int v = 0; v < 8; ++v) {
          const int m = v + 8 * hi;
          float z = fmaxf((acc[v] + bc) * sc + sh, 0.f);
          const int nl = (tile * 16 + m) / KNN_;
          atomicMax(&snode[nl * C2_ + col], __float_as_uint(z));
        }
      }
    }
  }
  __syncthreads();
  if (MODE == 0) {
    if (tid < C2_) { atomicAdd(&stats2[tid], ssum[tid]); atomicAdd(&stats2[C2_ + tid], sss[tid]); }
  } else {
    for (int q = tid; q < 16 * C2_; q += 128)
      node_out[(size_t)nodeBase * C2_ + q] = __uint_as_float(snode[q]);
  }
}

// ---------------------------------------------------------------------------
// Kernel 5: linear1 node[32768x128] @ W3[128x1024] via WMMA bf16.
// W3 chunk swizzled fragment-major in LDS (frag i = nt*4+ks).
// MODE 0: layer-3 stats.  MODE 1: BN3+ReLU + global max/sum pool per cloud.
// ---------------------------------------------------------------------------
template <int MODE>
__global__ void __launch_bounds__(128)
lin1_kernel(const float* __restrict__ node, const float* __restrict__ W3,
            const float* __restrict__ b3,
            const float* __restrict__ sc3, const float* __restrict__ sh3,
            float* __restrict__ stats3,
            unsigned* __restrict__ maxbuf, float* __restrict__ sumbuf) {
  __shared__ v16bf sw3v[64 * 32];                       // 64KB, fragment-major
  __shared__ bf16  sa[64 * C2_] __attribute__((aligned(16)));  // 16KB
  __shared__ float ssum[256], sss[256];
  const int tid = threadIdx.x;                          // blockDim = 128
  const int rowBase = blockIdx.x * 64;
  const int colBase = blockIdx.y * 256;
  {
    bf16* sw3b = (bf16*)sw3v;
    for (int q = tid; q < C2_ * 256; q += 128) {
      int k = q >> 8, c = q & 255;
      int ks = k >> 5, rem = k & 31, h2 = rem >> 4, e = rem & 15;
      int nt = c >> 4, n = c & 15;
      int L = h2 * 16 + n;
      sw3b[(((nt * 4 + ks) * 32 + L) << 4) + e] = (bf16)W3[(size_t)k * EMB_ + colBase + c];
    }
  }
  for (int q = tid; q < 64 * C2_; q += 128)
    sa[q] = (bf16)node[(size_t)rowBase * C2_ + q];
  if (MODE == 0 && tid < 128) { ssum[tid]=0.f; sss[tid]=0.f; ssum[tid+128]=0.f; sss[tid+128]=0.f; }
  __syncthreads();

  const int w = tid >> 5, lane = tid & 31, hi = lane >> 4, r = lane & 15;
  const bf16* arow = &sa[(w * 16 + r) * C2_];
  v16bf a[4];
#pragma unroll
  for (int ks = 0; ks < 4; ++ks) a[ks] = load_afrag(arow, hi, ks);
  const int bcloud = rowBase >> 11;

#pragma unroll 1
  for (int nt = 0; nt < 16; ++nt) {
    const int col = nt * 16 + r;
    v8f acc = {};
#pragma unroll
    for (int ks = 0; ks < 4; ++ks) {
      v16bf bb = sw3v[(nt * 4 + ks) * 32 + lane];
      acc = __builtin_amdgcn_wmma_f32_16x16x32_bf16(false, a[ks], false, bb, (short)0, acc, false, false);
    }
    const int cg = colBase + col;
    const float bc = b3[cg];
    if (MODE == 0) {
      float s = 0.f, ss = 0.f;
#pragma unroll
      for (int v = 0; v < 8; ++v) { float z = acc[v] + bc; s += z; ss += z * z; }
      atomicAdd(&ssum[col], s);
      atomicAdd(&sss[col], ss);
    } else {
      const float sc = sc3[cg], sh = sh3[cg];
      float mx = 0.f, s = 0.f;
#pragma unroll
      for (int v = 0; v < 8; ++v) {
        float z = fmaxf((acc[v] + bc) * sc + sh, 0.f);   // >= 0
        mx = fmaxf(mx, z); s += z;
      }
      atomicMax(&maxbuf[bcloud * EMB_ + cg], __float_as_uint(mx));
      atomicAdd(&sumbuf[bcloud * EMB_ + cg], s);
    }
  }
  if (MODE == 0) {
    __syncthreads();
    for (int q = tid; q < 256; q += 128) {
      atomicAdd(&stats3[colBase + q], ssum[q]);
      atomicAdd(&stats3[EMB_ + colBase + q], sss[q]);
    }
  }
}

// ---------------------------------------------------------------------------
// Kernel 6: pooled[b,0:1024]=max, [1024:2048]=mean; out = pooled@W4 + b4.
// ---------------------------------------------------------------------------
__global__ void final_kernel(const unsigned* __restrict__ maxbuf, const float* __restrict__ sumbuf,
                             const float* __restrict__ W4, const float* __restrict__ b4,
                             float* __restrict__ out) {
  __shared__ float red[256];
  const int b = blockIdx.x, t = threadIdx.x;
  float part[OUT_];
#pragma unroll
  for (int o = 0; o < OUT_; ++o) part[o] = 0.f;
  for (int c = t; c < EMB_; c += 256) {
    float mx = __uint_as_float(maxbuf[b * EMB_ + c]);
    float mn = sumbuf[b * EMB_ + c] * (1.0f / N_);
#pragma unroll
    for (int o = 0; o < OUT_; ++o)
      part[o] += mx * W4[(size_t)c * OUT_ + o] + mn * W4[(size_t)(EMB_ + c) * OUT_ + o];
  }
  for (int o = 0; o < OUT_; ++o) {
    red[t] = part[o];
    __syncthreads();
    for (int st = 128; st > 0; st >>= 1) { if (t < st) red[t] += red[t + st]; __syncthreads(); }
    if (t == 0) out[b * OUT_ + o] = red[0] + b4[o];
    __syncthreads();
  }
}

// ---------------------------------------------------------------------------
extern "C" void kernel_launch(void* const* d_in, const int* in_sizes, int n_in,
                              void* d_out, int out_size, void* d_ws, size_t ws_size,
                              hipStream_t stream) {
  (void)in_sizes; (void)n_in; (void)out_size; (void)ws_size;
  const float* pos = (const float*)d_in[0];
  const float* W1 = (const float*)d_in[1];  const float* b1 = (const float*)d_in[2];
  const float* g1 = (const float*)d_in[3];  const float* be1 = (const float*)d_in[4];
  const float* W2 = (const float*)d_in[5];  const float* b2 = (const float*)d_in[6];
  const float* g2 = (const float*)d_in[7];  const float* be2 = (const float*)d_in[8];
  const float* W3 = (const float*)d_in[9];  const float* b3 = (const float*)d_in[10];
  const float* g3 = (const float*)d_in[11]; const float* be3 = (const float*)d_in[12];
  const float* W4 = (const float*)d_in[13]; const float* b4 = (const float*)d_in[14];
  // d_in[15] is k (=20), hardcoded as KNN_.
  float* out = (float*)d_out;

  // ---- workspace layout (~20 MB) ----
  char* ws = (char*)d_ws;
  int*   idxb  = (int*)ws;        ws += (size_t)NEDGE_ * 4;
  float* node  = (float*)ws;      ws += (size_t)NROW_ * C2_ * 4;
  char*  zero0 = ws;
  float* stats1 = (float*)ws;     ws += 128 * 4;
  float* stats2 = (float*)ws;     ws += 256 * 4;
  float* stats3 = (float*)ws;     ws += 2048 * 4;
  unsigned* maxbuf = (unsigned*)ws; ws += (size_t)B_ * EMB_ * 4;
  float* sumbuf = (float*)ws;     ws += (size_t)B_ * EMB_ * 4;
  size_t zeroBytes = (size_t)(ws - zero0);
  float* sc1 = (float*)ws;  ws += 64 * 4;
  float* sh1 = (float*)ws;  ws += 64 * 4;
  float* sc2 = (float*)ws;  ws += 128 * 4;
  float* sh2 = (float*)ws;  ws += 128 * 4;
  float* sc3 = (float*)ws;  ws += 1024 * 4;
  float* sh3 = (float*)ws;  ws += 1024 * 4;

  hipMemsetAsync(zero0, 0, zeroBytes, stream);

  knn_kernel<<<dim3(N_ / 256, B_), 256, 0, stream>>>(pos, idxb);
  l1_stats_kernel<<<NEDGE_ / 64, 256, 0, stream>>>(pos, idxb, W1, b1, stats1);
  bn_finalize_kernel<<<1, 64, 0, stream>>>(stats1, g1, be1, sc1, sh1, 64, 1.0f / NEDGE_);
  edge_mlp_kernel<0><<<NROW_ / 16, 128, 0, stream>>>(pos, idxb, W1, b1, sc1, sh1,
                                                     W2, b2, nullptr, nullptr, stats2, nullptr);
  bn_finalize_kernel<<<1, 128, 0, stream>>>(stats2, g2, be2, sc2, sh2, 128, 1.0f / NEDGE_);
  edge_mlp_kernel<1><<<NROW_ / 16, 128, 0, stream>>>(pos, idxb, W1, b1, sc1, sh1,
                                                     W2, b2, sc2, sh2, nullptr, node);
  lin1_kernel<0><<<dim3(NROW_ / 64, 4), 128, 0, stream>>>(node, W3, b3, nullptr, nullptr,
                                                          stats3, nullptr, nullptr);
  bn_finalize_kernel<<<4, 256, 0, stream>>>(stats3, g3, be3, sc3, sh3, 1024, 1.0f / NROW_);
  lin1_kernel<1><<<dim3(NROW_ / 64, 4), 128, 0, stream>>>(node, W3, b3, sc3, sh3,
                                                          nullptr, maxbuf, sumbuf);
  final_kernel<<<B_, 256, 0, stream>>>(maxbuf, sumbuf, W4, b4, out);
}
